// AudioFeatures_86878598463494
// MI455X (gfx1250) — matrix operand order repaired and checked by reading, not compiled, not run on hardware
//
#include <hip/hip_runtime.h>
#include <hip/hip_bf16.h>

// ---------------------------------------------------------------------------
// AudioFeatures pipeline for MI455X (gfx1250, wave32, WMMA).
//
// Roofline reasoning: ~200 GFLOP total, dominated by 1x1 convs (= GEMMs with
// K in {64..1024}, N in {4096..524288}). We run all GEMMs on the matrix units
// via v_wmma_f32_16x16x32_bf16 (f32 accumulate) with bf16 channel-last
// operands so each WMMA fragment is two contiguous 16B loads per lane (per
// the CDNA5 ISA 16-bit A/B VGPR layout tables) and GEMM input traffic is
// halved vs f32 against the 23.3 TB/s HBM ceiling. Each wave owns a 16x64
// output tile (4 accumulators reusing one A fragment) -> 4 WMMAs per 10 b128
// loads. Elementwise / stencil / scan stages are VALU+LDS bandwidth work.
// ---------------------------------------------------------------------------

typedef __attribute__((ext_vector_type(16))) __bf16 v16bf;
typedef __attribute__((ext_vector_type(8)))  __bf16 v8bf;
typedef __attribute__((ext_vector_type(8)))  float  v8f;
typedef __attribute__((ext_vector_type(4)))  float  v4f;

#define SILU_GAIN 1.6778523489932886f           // 1/0.596
#define MPA       0.9191450300180579f           // 0.7/sqrt(0.58)
#define MPB       0.3939192985791676f           // 0.3/sqrt(0.58)
#define PIX_EPS   1e-4f

__device__ __forceinline__ float siluf(float x)    { return x / (1.f + __expf(-x)); }
__device__ __forceinline__ float mp_siluf(float x) { return SILU_GAIN * siluf(x); }
__device__ __forceinline__ float sigmf(float x)    { return 1.f / (1.f + __expf(-x)); }

// ---------------------------------------------------------------------------
// WMMA fragment loaders (CDNA5 ISA §7.12.2, 16-bit 16x16x32).
// A (MxK=16x32, row-major [m][k]):
//   lane L<16: row m0+L, k-chunks {k0+0..7} -> v0..7, {k0+16..23} -> v8..15
//   lane L>=16: row m0+L-16, chunks {k0+8..15}, {k0+24..31}
// B (KxN=32x16) stored transposed (activations channel-last, [n][k]):
//   lane L<16: col n0+L, k {k0+0..7},{k0+16..23}; lane L>=16: col n0+L-16, k {k0+8..15},{k0+24..31}
// ---------------------------------------------------------------------------
__device__ __forceinline__ v16bf load_frag_a(const __bf16* __restrict__ A, int K,
                                             int m0, int k0, int lane) {
  int row  = m0 + (lane & 15);
  int koff = (lane >> 4) * 8;
  const __bf16* p = A + (size_t)row * K + k0 + koff;
  union { v16bf v; v8bf h[2]; } u;
  u.h[0] = *(const v8bf*)(p);
  u.h[1] = *(const v8bf*)(p + 16);
  return u.v;
}

__device__ __forceinline__ v16bf load_frag_b(const __bf16* __restrict__ Bm, int K,
                                             int n0, int k0, int lane) {
  int col  = n0 + (lane & 15);
  int koff = (lane >> 4) * 8;
  const __bf16* p = Bm + (size_t)col * K + k0 + koff;
  union { v16bf v; v8bf h[2]; } u;
  u.h[0] = *(const v8bf*)(p);
  u.h[1] = *(const v8bf*)(p + 16);
  return u.v;
}

__device__ __forceinline__ void store_acc(float* __restrict__ C, int M,
                                          int m0, int n, int hi, v8f acc) {
  // C/D layout: VGPR r -> M = m0 + r + 8*hi (lanes>=16 are hi half), col = n.
  float* p = C + (size_t)n * M + m0 + hi * 8;
  v4f lo = {acc[0], acc[1], acc[2], acc[3]};
  v4f hh = {acc[4], acc[5], acc[6], acc[7]};
  *(v4f*)p       = lo;
  *(v4f*)(p + 4) = hh;
}

// ---------------------------------------------------------------------------
// GEMM: C[n][m] (f32, channel-last) = sum_k A[m][k] * Bact[n][k], bf16 inputs.
// Block = 8 waves; each wave owns a 16(M) x 64(N) tile (4 accumulators, one
// shared A fragment) and loops K in steps of 32. Block covers 512 columns.
// ---------------------------------------------------------------------------
__global__ __launch_bounds__(256)
void k_gemm_bf16(const __bf16* __restrict__ A, const __bf16* __restrict__ Bm,
                 float* __restrict__ C, int M, int K, int N) {
  int lane = threadIdx.x & 31;
  int wave = threadIdx.x >> 5;
  int n0   = blockIdx.x * 512 + wave * 64;
  int m0   = blockIdx.y * 16;
  v8f acc0 = {};
  v8f acc1 = {};
  v8f acc2 = {};
  v8f acc3 = {};
  for (int k0 = 0; k0 < K; k0 += 32) {
    v16bf a  = load_frag_a(A, K, m0, k0, lane);
    v16bf b0 = load_frag_b(Bm, K, n0,      k0, lane);
    v16bf b1 = load_frag_b(Bm, K, n0 + 16, k0, lane);
    v16bf b2 = load_frag_b(Bm, K, n0 + 32, k0, lane);
    v16bf b3 = load_frag_b(Bm, K, n0 + 48, k0, lane);
    acc0 = __builtin_amdgcn_wmma_f32_16x16x32_bf16(false, a, false, b0, (short)0, acc0, false, false);
    acc1 = __builtin_amdgcn_wmma_f32_16x16x32_bf16(false, a, false, b1, (short)0, acc1, false, false);
    acc2 = __builtin_amdgcn_wmma_f32_16x16x32_bf16(false, a, false, b2, (short)0, acc2, false, false);
    acc3 = __builtin_amdgcn_wmma_f32_16x16x32_bf16(false, a, false, b3, (short)0, acc3, false, false);
  }
  int col = lane & 15;
  int hi  = lane >> 4;
  store_acc(C, M, m0, n0 + col,      hi, acc0);
  store_acc(C, M, m0, n0 + 16 + col, hi, acc1);
  store_acc(C, M, m0, n0 + 32 + col, hi, acc2);
  store_acc(C, M, m0, n0 + 48 + col, hi, acc3);
}

// ---------------------------------------------------------------------------
// Weight prep: norm_w (per output row L2 norm) -> bf16; depthwise stays f32.
// ---------------------------------------------------------------------------
__global__ __launch_bounds__(256)
void k_normw_row(const float* __restrict__ src, __bf16* __restrict__ dst, int K) {
  int m = blockIdx.x, t = threadIdx.x;
  __shared__ float red[256];
  const float* row = src + (size_t)m * K;
  float s = 0.f;
  for (int k = t; k < K; k += 256) { float w = row[k]; s += w * w; }
  red[t] = s; __syncthreads();
  for (int d = 128; d > 0; d >>= 1) { if (t < d) red[t] += red[t + d]; __syncthreads(); }
  float rinv = rsqrtf(red[0] + 1e-8f);
  for (int k = t; k < K; k += 256) dst[(size_t)m * K + k] = (__bf16)(row[k] * rinv);
}

__global__ __launch_bounds__(256)
void k_normw_dw(const float* __restrict__ src, float* __restrict__ dst,
                int taps, int chans) {
  int c = blockIdx.x * blockDim.x + threadIdx.x;
  if (c >= chans) return;
  float s = 0.f;
  for (int t = 0; t < taps; t++) { float w = src[c * taps + t]; s += w * w; }
  float rinv = rsqrtf(s + 1e-8f);
  for (int t = 0; t < taps; t++) dst[c * taps + t] = src[c * taps + t] * rinv;
}

// ---------------------------------------------------------------------------
// proj: X[b][p][c] = audio[b][p] * pw[c] + pb[c]   (channel-last, C=64)
// ---------------------------------------------------------------------------
__global__ __launch_bounds__(256)
void k_proj(const float* __restrict__ audio, const float* __restrict__ pw,
            const float* __restrict__ pb, float* __restrict__ X, long total) {
  long idx = (long)blockIdx.x * blockDim.x + threadIdx.x;
  if (idx >= total) return;
  int c = (int)(idx & 63);
  X[idx] = audio[idx >> 6] * pw[c] + pb[c];
}

// ---------------------------------------------------------------------------
// pixel_norm (+ optional mp_silu) -> bf16 activation; invr saved for residual.
// One thread per position; channels contiguous (channel-last).
// ---------------------------------------------------------------------------
__global__ __launch_bounds__(256)
void k_pnorm(const float* __restrict__ X, float* __restrict__ invr,
             __bf16* __restrict__ act, int C, int Np, int apply_silu) {
  int n = blockIdx.x * blockDim.x + threadIdx.x;
  if (n >= Np) return;
  const float* row = X + (size_t)n * C;
  float ss = 0.f;
  for (int c = 0; c < C; c += 4) {
    v4f v = *(const v4f*)(row + c);
    ss += v[0]*v[0] + v[1]*v[1] + v[2]*v[2] + v[3]*v[3];
  }
  float r = rsqrtf(ss / (float)C + PIX_EPS);
  invr[n] = r;
  __bf16* out = act + (size_t)n * C;
  for (int c = 0; c < C; c += 4) {
    v4f v = *(const v4f*)(row + c);
#pragma unroll
    for (int j = 0; j < 4; j++) {
      float x = v[j] * r;
      if (apply_silu) x = mp_siluf(x);
      out[c + j] = (__bf16)x;
    }
  }
}

// ---------------------------------------------------------------------------
// depthwise 5x3 conv over (F,L), zero pad (2,2),(1,1), then mp_silu -> bf16.
// ---------------------------------------------------------------------------
__global__ __launch_bounds__(256)
void k_dw2d(const float* __restrict__ R1, const float* __restrict__ W,
            __bf16* __restrict__ out, int C, int F, int L, long total) {
  long idx = (long)blockIdx.x * blockDim.x + threadIdx.x;
  if (idx >= total) return;
  int  c   = (int)(idx % C);
  long pos = idx / C;
  int  l   = (int)(pos % L);
  long bf  = pos / L;
  int  f   = (int)(bf % F);
  long b   = bf / F;
  float acc = 0.f;
  const float* w = W + (size_t)c * 15;
#pragma unroll
  for (int kf = 0; kf < 5; kf++) {
    int ff = f + kf - 2;
    if (ff < 0 || ff >= F) continue;
#pragma unroll
    for (int kl = 0; kl < 3; kl++) {
      int ll = l + kl - 1;
      if (ll < 0 || ll >= L) continue;
      acc += R1[(((b * F + ff) * L) + ll) * C + c] * w[kf * 3 + kl];
    }
  }
  out[idx] = (__bf16)mp_siluf(acc);
}

// ---------------------------------------------------------------------------
// h = mp_add(pixel_norm(x), r); avgpool over freq by s -> bf16.
// ---------------------------------------------------------------------------
__global__ __launch_bounds__(256)
void k_addpool(const float* __restrict__ X, const float* __restrict__ invr,
               const float* __restrict__ R2, __bf16* __restrict__ out,
               int C, int F, int s, int L, long total) {
  long idx = (long)blockIdx.x * blockDim.x + threadIdx.x;
  if (idx >= total) return;
  int  c   = (int)(idx % C);
  long pos = idx / C;
  int  l   = (int)(pos % L);
  long rem = pos / L;
  int  F2  = F / s;
  int  f2  = (int)(rem % F2);
  long b   = rem / F2;
  float acc = 0.f;
  for (int j = 0; j < s; j++) {
    int  f = f2 * s + j;
    long n = (b * F + f) * L + l;
    float xn = X[n * C + c] * invr[n];
    acc += MPA * xn + MPB * R2[n * C + c];
  }
  out[idx] = (__bf16)(acc / (float)s);
}

// ---------------------------------------------------------------------------
// seq: mp_silu(h), depthwise conv1d k=3 pad 1, mp_silu -> bf16.
// HG is [n][Ctot]; h occupies channels [0, Ch).
// ---------------------------------------------------------------------------
__global__ __launch_bounds__(256)
void k_dw1d(const float* __restrict__ HG, const float* __restrict__ W,
            __bf16* __restrict__ out, int L, int Ctot, int Ch, long total) {
  long idx = (long)blockIdx.x * blockDim.x + threadIdx.x;
  if (idx >= total) return;
  int  c   = (int)(idx % Ch);
  long pos = idx / Ch;
  int  l   = (int)(pos % L);
  long b   = pos / L;
  float acc = 0.f;
#pragma unroll
  for (int k = 0; k < 3; k++) {
    int ll = l + k - 1;
    if (ll < 0 || ll >= L) continue;
    float h = HG[((b * L + ll) * Ctot) + c];
    acc += mp_siluf(h) * W[c * 3 + k];
  }
  out[(b * L + l) * Ch + c] = (__bf16)mp_siluf(acc);
}

// ---------------------------------------------------------------------------
// min_gru scan + gate. One block per (batch, channel); L = 2048 = 256 thr x 8.
// comb((a1,b1),(a2,b2)) = (a2*a1, a2*b1 + b2); LDS Hillis-Steele over 256
// composed segments, then local replay; output *= mp_silu(g), store bf16.
// ---------------------------------------------------------------------------
__global__ __launch_bounds__(256)
void k_scan(const float* __restrict__ G, const float* __restrict__ HG,
            __bf16* __restrict__ SB, int L, int Cz, int Ctot) {
  int blk = blockIdx.x;
  int ch  = blk % Cz;
  int b   = blk / Cz;
  int t   = threadIdx.x;
  const int E = 8;
  float av[E], bv[E];
  float Ac = 1.f, Bc = 0.f;
  int l0 = t * E;
  for (int i = 0; i < E; i++) {
    long n = (long)(b * L + l0 + i) * Ctot;
    float z  = sigmf(G[n + ch]);
    float cc = G[n + Cz + ch];
    float a  = 1.f - z;
    float bb = z * cc;
    av[i] = a; bv[i] = bb;
    Ac = a * Ac;
    Bc = a * Bc + bb;
  }
  __shared__ float sA[256];
  __shared__ float sB[256];
  sA[t] = Ac; sB[t] = Bc;
  __syncthreads();
  for (int d = 1; d < 256; d <<= 1) {
    float pa = 0.f, pb = 0.f, ca = 0.f, cb = 0.f;
    if (t >= d) { pa = sA[t - d]; pb = sB[t - d]; ca = sA[t]; cb = sB[t]; }
    __syncthreads();
    if (t >= d) { sA[t] = ca * pa; sB[t] = ca * pb + cb; }
    __syncthreads();
  }
  float carry = (t == 0) ? 0.f : sB[t - 1];
  for (int i = 0; i < E; i++) {
    carry = av[i] * carry + bv[i];
    long n = (long)(b * L + l0 + i) * Ctot;
    float g = HG[n + Cz + ch];
    SB[(long)(b * L + l0 + i) * Cz + ch] = (__bf16)(carry * mp_siluf(g));
  }
}

// ---------------------------------------------------------------------------
// X = mp_add(X * invr, R)   (in-place, elementwise)
// ---------------------------------------------------------------------------
__global__ __launch_bounds__(256)
void k_mpadd(float* __restrict__ X, const float* __restrict__ invr,
             const float* __restrict__ R, long total, int C) {
  long idx = (long)blockIdx.x * blockDim.x + threadIdx.x;
  if (idx >= total) return;
  long n = idx / C;
  float xn = X[idx] * invr[n];
  X[idx] = MPA * xn + MPB * R[idx];
}

// ---------------------------------------------------------------------------
// final: out[b][c][l] = mp_silu(X[b][l][c])  (transpose back to NCL, f32)
// ---------------------------------------------------------------------------
__global__ __launch_bounds__(256)
void k_final(const float* __restrict__ X, float* __restrict__ out,
             int C, int L, long total) {
  long idx = (long)blockIdx.x * blockDim.x + threadIdx.x;
  if (idx >= total) return;
  long b   = idx / ((long)C * L);
  long rem = idx % ((long)C * L);
  int  c   = (int)(rem / L);
  int  l   = (int)(rem % L);
  out[idx] = mp_siluf(X[(b * L + l) * (long)C + c]);
}

// ---------------------------------------------------------------------------
// Host orchestration
// ---------------------------------------------------------------------------
static void gemm(const __bf16* A, const __bf16* Bm, float* C, int M, int K, int N,
                 hipStream_t s) {
  dim3 g((unsigned)(N / 512), (unsigned)(M / 16));
  k_gemm_bf16<<<g, 256, 0, s>>>(A, Bm, C, M, K, N);
}

extern "C" void kernel_launch(void* const* d_in, const int* in_sizes, int n_in,
                              void* d_out, int out_size, void* d_ws, size_t ws_size,
                              hipStream_t stream) {
  (void)in_sizes; (void)n_in; (void)out_size; (void)ws_size;

  const float* audio  = (const float*)d_in[0];
  const float* proj_w = (const float*)d_in[1];
  const float* proj_b = (const float*)d_in[2];
  const float* cw1[3] = {(const float*)d_in[3],  (const float*)d_in[7],  (const float*)d_in[11]};
  const float* cwd[3] = {(const float*)d_in[4],  (const float*)d_in[8],  (const float*)d_in[12]};
  const float* cw2[3] = {(const float*)d_in[5],  (const float*)d_in[9],  (const float*)d_in[13]};
  const float* cdn[3] = {(const float*)d_in[6],  (const float*)d_in[10], (const float*)d_in[14]};
  const float* hg_w   = (const float*)d_in[15];
  const float* dw_w   = (const float*)d_in[16];
  const float* gru_w  = (const float*)d_in[17];
  const float* out_w  = (const float*)d_in[18];

  const int B = 2, L = 2048;
  const int Cs[3]  = {64, 128, 256};
  const int C2s[3] = {128, 256, 512};
  const int Ss[3]  = {4, 4, 8};

  // ---- workspace carve (deterministic every call) ----
  char*  ws  = (char*)d_ws;
  size_t off = 0;
  auto carve = [&](size_t bytes) -> void* {
    void* p = ws + off;
    off = (off + bytes + 255) & ~(size_t)255;
    return p;
  };

  __bf16* wb1[3];  __bf16* wb2[3];  __bf16* wbdn[3];  float* wdw2[3];
  for (int i = 0; i < 3; i++) {
    wb1[i]  = (__bf16*)carve((size_t)C2s[i] * Cs[i]  * 2);
    wb2[i]  = (__bf16*)carve((size_t)Cs[i]  * C2s[i] * 2);
    wbdn[i] = (__bf16*)carve((size_t)C2s[i] * Cs[i]  * 2);
    wdw2[i] = (float*)carve((size_t)C2s[i] * 15 * 4);
  }
  __bf16* whg  = (__bf16*)carve((size_t)4 * 2048 * 512  * 2);
  __bf16* wgru = (__bf16*)carve((size_t)4 * 2048 * 1024 * 2);
  __bf16* wout = (__bf16*)carve((size_t)4 * 512  * 1024 * 2);
  float*  wdw1 = (float*)carve((size_t)4 * 1024 * 3 * 4);
  float*  invr = (float*)carve((size_t)B * 128 * L * 4);          // 2 MB
  float*  T0   = (float*)carve((size_t)B * 128 * L * 128 * 4);    // 256 MB (widest f32)
  float*  T1   = (float*)carve((size_t)B * 128 * L * 128 * 2);    // 128 MB (bf16/f32 pool)
  float*  T2   = (float*)carve((size_t)B * 128 * L * 64  * 4);    // 128 MB (residual X)
  __bf16* T3   = (__bf16*)carve((size_t)B * 128 * L * 64 * 2);    // 64 MB (act bf16)
  __bf16* T1b  = (__bf16*)T1;
  float*  Gbuf = (float*)((char*)T0 + ((size_t)64  << 20));
  float*  OUTC = (float*)((char*)T0 + ((size_t)128 << 20));

  // ---- weight normalization (norm_w) ----
  for (int i = 0; i < 3; i++) {
    k_normw_row<<<C2s[i], 256, 0, stream>>>(cw1[i], wb1[i],  Cs[i]);
    k_normw_row<<<Cs[i],  256, 0, stream>>>(cw2[i], wb2[i],  C2s[i]);
    k_normw_row<<<C2s[i], 256, 0, stream>>>(cdn[i], wbdn[i], Cs[i]);
    k_normw_dw<<<(C2s[i] + 255) / 256, 256, 0, stream>>>(cwd[i], wdw2[i], 15, C2s[i]);
  }
  k_normw_row<<<4 * 2048, 256, 0, stream>>>(hg_w,  whg,  512);
  k_normw_row<<<4 * 2048, 256, 0, stream>>>(gru_w, wgru, 1024);
  k_normw_row<<<4 * 512,  256, 0, stream>>>(out_w, wout, 1024);
  k_normw_dw<<<(4096 + 255) / 256, 256, 0, stream>>>(dw_w, wdw1, 3, 4096);

  // ---- proj: audio (B,128,L) -> X channel-last (B, 128*L, 64) ----
  long totX = (long)B * 128 * L * 64;
  k_proj<<<(unsigned)(totX / 256), 256, 0, stream>>>(audio, proj_w, proj_b, T2, totX);

  // ---- conv blocks ----
  int F = 128;
  for (int i = 0; i < 3; i++) {
    int C = Cs[i], C2 = C2s[i], s = Ss[i];
    long Np = (long)B * F * L;
    k_pnorm<<<(unsigned)(Np / 256), 256, 0, stream>>>(T2, invr, T3, C, (int)Np, 1);
    gemm(wb1[i], T3, T0, C2, C, (int)Np, stream);                 // conv1 (WMMA)
    long tdw = Np * C2;
    k_dw2d<<<(unsigned)(tdw / 256), 256, 0, stream>>>(T0, wdw2[i], T1b, C2, F, L, tdw);
    gemm(wb2[i], T1b, T0, C, C2, (int)Np, stream);                // conv2 (WMMA)
    int  F2  = F / s;
    long Np2 = (long)B * F2 * L;
    long tap = Np2 * C;
    k_addpool<<<(unsigned)(tap / 256), 256, 0, stream>>>(T2, invr, T0, T3, C, F, s, L, tap);
    gemm(wbdn[i], T3, T2, C2, C, (int)Np2, stream);               // downsample conv (WMMA)
    F = F2;
  }

  // ---- seq blocks: X is (B, L, 512) channel-last in T2 ----
  const int Npos = B * L;  // 4096
  for (int i = 0; i < 4; i++) {
    k_pnorm<<<(unsigned)(Npos / 256), 256, 0, stream>>>(T2, invr, T3, 512, Npos, 0);
    gemm(whg + (size_t)i * 2048 * 512, T3, T0, 2048, 512, Npos, stream);   // hg (WMMA)
    long th = (long)Npos * 1024;
    k_dw1d<<<(unsigned)(th / 256), 256, 0, stream>>>(T0, wdw1 + (size_t)i * 1024 * 3,
                                                     T1b, L, 2048, 1024, th);
    gemm(wgru + (size_t)i * 2048 * 1024, T1b, Gbuf, 2048, 1024, Npos, stream); // gru (WMMA)
    k_scan<<<B * 1024, 256, 0, stream>>>(Gbuf, T0, T1b, L, 1024, 2048);
    gemm(wout + (size_t)i * 512 * 1024, T1b, OUTC, 512, 1024, Npos, stream);   // out (WMMA)
    long ta = (long)Npos * 512;
    k_mpadd<<<(unsigned)(ta / 256), 256, 0, stream>>>(T2, invr, OUTC, ta, 512);
  }

  // ---- final mp_silu + transpose to (B, 512, L) f32 ----
  long tout = (long)B * 512 * L;
  k_final<<<(unsigned)(tout / 256), 256, 0, stream>>>(T2, (float*)d_out, 512, L, tout);
}
